// GraphSAGE_49297634624094
// MI455X (gfx1250) — compile-verified, matrix operand order checked by reading, not run on hardware
//
#include <hip/hip_runtime.h>

#define N_NODES 50000
#define N_EDGES 800000
#define IN_CH   256
#define HID     512

typedef __attribute__((ext_vector_type(16))) __bf16 v16bf;
typedef __attribute__((ext_vector_type(8)))  __bf16 v8bf;
typedef __attribute__((ext_vector_type(8)))  float  v8f;

// ---------- fp32 -> bf16 (round-to-nearest-even) ----------
__device__ __forceinline__ unsigned short f2bf(float f) {
  unsigned u = __float_as_uint(f);
  if ((u & 0x7fffffffu) > 0x7f800000u) return (unsigned short)((u >> 16) | 0x0040u); // quiet NaN
  unsigned r = u + 0x7fffu + ((u >> 16) & 1u);
  return (unsigned short)(r >> 16);
}

// ---------- weight pack: W[din][dout] f32 -> Wt[dout][din] bf16 ----------
__global__ void pack_weights(const float* __restrict__ W, unsigned short* __restrict__ Wt,
                             int din, int dout) {
  long i = (long)blockIdx.x * blockDim.x + threadIdx.x;
  long n = (long)din * dout;
  if (i >= n) return;
  int r = (int)(i / dout);
  int c = (int)(i % dout);
  Wt[(long)c * din + r] = f2bf(W[i]);
}

// ---------- elementwise f32 -> bf16 buffer ----------
__global__ void convert_bf16(const float* __restrict__ in, unsigned short* __restrict__ out, long n) {
  long i = (long)blockIdx.x * blockDim.x + threadIdx.x;
  if (i < n) out[i] = f2bf(in[i]);
}

// ---------- edge scatter: agg[dst] += feat[src]  (fp32 atomics, 4 ch / thread) ----------
__global__ void scatter_add(const float* __restrict__ feat,
                            const int* __restrict__ src, const int* __restrict__ dst,
                            float* __restrict__ agg, int din, int gshift) {
  long tid = (long)blockIdx.x * blockDim.x + threadIdx.x;
  long total = (long)N_EDGES << gshift;          // groups of 4 channels per edge
  if (tid >= total) return;
  int e = (int)(tid >> gshift);
  int g = (int)(tid & ((1 << gshift) - 1)) << 2; // channel base
  int s = src[e];
  int d = dst[e];
  const float4 v = *(const float4*)(feat + (long)s * din + g);
  float* p = agg + (long)d * din + g;
  atomicAdd(p + 0, v.x);
  atomicAdd(p + 1, v.y);
  atomicAdd(p + 2, v.z);
  atomicAdd(p + 3, v.w);
}

// ---------- fragment load: ISA 16-bit A/B layout (two 16B chunks, K and K+16) ----------
__device__ __forceinline__ v16bf load_frag(const __bf16* p) {
  v8bf lo = *(const v8bf*)(p);        // K = k .. k+7   (for this lane's half)
  v8bf hi = *(const v8bf*)(p + 16);   // K = k+16 .. k+23
  v16bf r;
#pragma unroll
  for (int i = 0; i < 8; ++i) { r[i] = lo[i]; r[i + 8] = hi[i]; }
  return r;
}

// ---------- fused SAGE GEMM: out = agg_bf16 @ WlT + feat_bf16 @ WrT + bias ----------
// One wave computes a 16x64 output slab (1 M-tile x 4 N-tiles). Per k-step:
// batch-issue all 20 b128 loads (A pair + 8 B fragments in distinct registers),
// then drain 8 WMMAs -> deep load/WMMA pipelining instead of wait-per-WMMA.
__global__ void sage_gemm(const __bf16* __restrict__ Aagg, const __bf16* __restrict__ Afeat,
                          const __bf16* __restrict__ WlT,  const __bf16* __restrict__ WrT,
                          const float* __restrict__ bias,  float* __restrict__ out,
                          int din, int dout) {
  const int wave = threadIdx.x >> 5;
  const int lane = threadIdx.x & 31;
  const int ngrp = dout >> 6;                       // N-groups of 4 tiles (64 cols) = 8
  int task = blockIdx.x * (blockDim.x >> 5) + wave; // (N_NODES/16) * ngrp tasks
  if (task >= (N_NODES / 16) * ngrp) return;
  const int tm  = task / ngrp;                      // M-tile (same for all waves in block)
  const int tnb = (task - tm * ngrp) * 4;           // first of 4 N-tiles

  const int half = lane >> 4;          // 0: K lo-half, 1: K hi-half per ISA layout
  const int lr   = lane & 15;
  const long arow  = (long)(tm * 16 + lr) * din;    // A row for this lane (M = lr)
  const long brow0 = (long)(tnb * 16 + lr) * din;   // B row = output column (N = lr)
  const long bstp  = (long)16 * din;                // stride between N-tiles in packed W
  const int  koff  = half * 8;

  v8f acc[4] = {v8f{}, v8f{}, v8f{}, v8f{}};
  for (int k0 = 0; k0 < din; k0 += 32) {
    const __bf16* pa1 = Aagg  + arow + k0 + koff;
    const __bf16* pa2 = Afeat + arow + k0 + koff;
    __builtin_prefetch(pa1 + 32, 0, 3);             // global_prefetch_b8: next k-chunk
    __builtin_prefetch(pa2 + 32, 0, 3);

    // ---- issue ALL loads for this k-step first (distinct registers) ----
    v16bf a1 = load_frag(pa1);                      // shared across 4 N-tiles
    v16bf a2 = load_frag(pa2);
    v16bf bL[4], bR[4];
#pragma unroll
    for (int j = 0; j < 4; ++j) {
      bL[j] = load_frag(WlT + brow0 + j * bstp + k0 + koff);
      bR[j] = load_frag(WrT + brow0 + j * bstp + k0 + koff);
    }

    // ---- then drain 8 WMMAs against the in-flight loads ----
#pragma unroll
    for (int j = 0; j < 4; ++j) {
      acc[j] = __builtin_amdgcn_wmma_f32_16x16x32_bf16(false, a1, false, bL[j],
                                                       (short)0, acc[j], false, false);
      acc[j] = __builtin_amdgcn_wmma_f32_16x16x32_bf16(false, a2, false, bR[j],
                                                       (short)0, acc[j], false, false);
    }
  }

  // C/D layout: VGPR r -> M = r (lanes 0-15) / M = r+8 (lanes 16-31); N = lane&15
  const int rbase = tm * 16 + half * 8;
#pragma unroll
  for (int j = 0; j < 4; ++j) {
    const int col  = (tnb + j) * 16 + lr;
    const float bv = bias[col];
#pragma unroll
    for (int r = 0; r < 8; ++r)
      out[(long)(rbase + r) * dout + col] = acc[j][r] + bv;
  }
}

// ---------- per-row L2 normalize (eps=1e-12) + ELU ----------
__device__ __forceinline__ float wave_red(float v) {
#pragma unroll
  for (int m = 16; m > 0; m >>= 1) v += __shfl_xor(v, m, 32);
  return v;
}

__global__ void normalize_elu(const float* in, float* out, int ch) {
  const int row  = blockIdx.x;
  const int wave = threadIdx.x >> 5;
  const int lane = threadIdx.x & 31;
  const float* r = in + (long)row * ch;
  float ss = 0.f;
  for (int c = threadIdx.x; c < ch; c += blockDim.x) { float v = r[c]; ss += v * v; }
  ss = wave_red(ss);
  __shared__ float smem[8];
  if (lane == 0) smem[wave] = ss;
  __syncthreads();
  if (wave == 0) {
    float t = (lane < (int)(blockDim.x >> 5)) ? smem[lane] : 0.f;
    t = wave_red(t);
    if (lane == 0) smem[0] = t;
  }
  __syncthreads();
  const float inv = 1.0f / fmaxf(sqrtf(smem[0]), 1e-12f);
  float* w = out + (long)row * ch;
  for (int c = threadIdx.x; c < ch; c += blockDim.x) {
    float v = r[c] * inv;
    w[c] = (v > 0.f) ? v : expm1f(v);   // ELU(alpha=1)
  }
}

// ============================ host launcher ============================
extern "C" void kernel_launch(void* const* d_in, const int* in_sizes, int n_in,
                              void* d_out, int out_size, void* d_ws, size_t ws_size,
                              hipStream_t stream) {
  const float* x   = (const float*)d_in[0];
  const int*   ei  = (const int*)d_in[1];
  const int*   src = ei;
  const int*   dst = ei + N_EDGES;
  const float* Wl[3] = { (const float*)d_in[2], (const float*)d_in[5], (const float*)d_in[8]  };
  const float* bl[3] = { (const float*)d_in[3], (const float*)d_in[6], (const float*)d_in[9]  };
  const float* Wr[3] = { (const float*)d_in[4], (const float*)d_in[7], (const float*)d_in[10] };
  const int din_l[3] = { IN_CH, HID, HID };

  // ---- carve workspace ----
  char* ws = (char*)d_ws;
  size_t off = 0;
  auto carve = [&](size_t bytes) -> void* {
    void* p = ws + off;
    off += (bytes + 255) & ~(size_t)255;
    return p;
  };
  float*          agg  = (float*)         carve((size_t)N_NODES * HID * 4);
  unsigned short* aggb = (unsigned short*)carve((size_t)N_NODES * HID * 2);
  unsigned short* fb   = (unsigned short*)carve((size_t)N_NODES * HID * 2);
  float*          buf0 = (float*)         carve((size_t)N_NODES * HID * 4);
  float*          buf1 = (float*)         carve((size_t)N_NODES * HID * 4);
  unsigned short* WlT[3], *WrT[3];
  for (int l = 0; l < 3; ++l) {
    WlT[l] = (unsigned short*)carve((size_t)din_l[l] * HID * 2);
    WrT[l] = (unsigned short*)carve((size_t)din_l[l] * HID * 2);
  }

  // ---- pack weights (transpose + bf16) ----
  for (int l = 0; l < 3; ++l) {
    long n = (long)din_l[l] * HID;
    int blocks = (int)((n + 255) / 256);
    pack_weights<<<blocks, 256, 0, stream>>>(Wl[l], WlT[l], din_l[l], HID);
    pack_weights<<<blocks, 256, 0, stream>>>(Wr[l], WrT[l], din_l[l], HID);
  }

  // ---- layer schedule: feat -> (scatter, gemm, normalize+elu) ----
  const float* feat = x;
  for (int l = 0; l < 3; ++l) {
    const int din = din_l[l];
    const long nf = (long)N_NODES * din;
    const int gshift = (din == IN_CH) ? 6 : 7;   // din/4 groups per edge

    hipMemsetAsync(agg, 0, (size_t)nf * 4, stream);

    convert_bf16<<<(int)((nf + 255) / 256), 256, 0, stream>>>(feat, fb, nf);

    long sc_total = (long)N_EDGES << gshift;
    scatter_add<<<(int)((sc_total + 255) / 256), 256, 0, stream>>>(feat, src, dst, agg, din, gshift);

    convert_bf16<<<(int)((nf + 255) / 256), 256, 0, stream>>>(agg, aggb, nf);

    float* raw      = (l == 2) ? buf1 : buf0;           // l0: buf0, l1: buf0, l2: buf1
    float* feat_out = (l == 0) ? buf1 : (l == 1) ? buf0 : (float*)d_out;

    int tasks  = (N_NODES / 16) * (HID / 64);           // 3125 * 8 = 25000 wave-tasks
    int blocks = tasks / 8;                             // 8 waves per 256-thread block
    sage_gemm<<<blocks, 256, 0, stream>>>((const __bf16*)aggb, (const __bf16*)fb,
                                          (const __bf16*)WlT[l], (const __bf16*)WrT[l],
                                          bl[l], raw, din, HID);

    normalize_elu<<<N_NODES, 256, 0, stream>>>(raw, feat_out, HID);
    feat = feat_out;
  }
}
